// MultiHeadAttention_78262894068012
// MI455X (gfx1250) — matrix operand order repaired and checked by reading, not compiled, not run on hardware
//
#include <hip/hip_runtime.h>
#include <hip/hip_bf16.h>
#include <math.h>

#define B_  2
#define S_  2048
#define D_  1024
#define H_  16
#define HD_ 64
#define SCALE_ 0.03125f   // D^-0.5 = 1/32

typedef __bf16 bf16_t;
typedef __attribute__((ext_vector_type(16))) __bf16 v16bf;
typedef __attribute__((ext_vector_type(8)))  __bf16 v8bf;
typedef __attribute__((ext_vector_type(8)))  float  v8f;

static_assert(H_ * HD_ == D_, "head layout");

__device__ __forceinline__ bf16_t f32_to_bf16(float f) {
  unsigned int u = __builtin_bit_cast(unsigned int, f);
  u += 0x7FFFu + ((u >> 16) & 1u);          // round-to-nearest-even
  unsigned short h = (unsigned short)(u >> 16);
  return __builtin_bit_cast(bf16_t, h);
}

// ---- WMMA fragment loaders (wave32, 16x16x32 bf16 layouts, ISA 7.12.2) ----
// A matrix 16x32 (MxK), row-major source with leading dim ld.
// lane<16: row=lane,    K = k0+{0..7} U k0+{16..23}
// lane>=16: row=lane-16, K = k0+{8..15} U k0+{24..31}
__device__ __forceinline__ v16bf load_A(const bf16_t* __restrict__ src, int ld,
                                        int row0, int k0) {
  const int lane = threadIdx.x & 31;
  const int r  = row0 + (lane & 15);
  const int kb = k0 + ((lane & 16) ? 8 : 0);
  const bf16_t* p = src + (size_t)r * ld + kb;
  v8bf lo = *reinterpret_cast<const v8bf*>(p);
  v8bf hi = *reinterpret_cast<const v8bf*>(p + 16);
  v16bf a;
#pragma unroll
  for (int i = 0; i < 8; ++i) { a[i] = lo[i]; a[8 + i] = hi[i]; }
  return a;
}

// B matrix 32x16 (KxN) loaded from BT (N-major: BT[n][k], leading dim ldk).
// lane<16: col=lane,    K = k0+{0..7} U k0+{16..23} ; lane>=16: col=lane-16, K = k0+{8..15} U k0+{24..31}
__device__ __forceinline__ v16bf load_B(const bf16_t* __restrict__ bt, int ldk,
                                        int n0, int k0) {
  const int lane = threadIdx.x & 31;
  const int n  = n0 + (lane & 15);
  const int kb = k0 + ((lane & 16) ? 8 : 0);
  const bf16_t* p = bt + (size_t)n * ldk + kb;
  v8bf lo = *reinterpret_cast<const v8bf*>(p);
  v8bf hi = *reinterpret_cast<const v8bf*>(p + 16);
  v16bf b;
#pragma unroll
  for (int i = 0; i < 8; ++i) { b[i] = lo[i]; b[8 + i] = hi[i]; }
  return b;
}

__device__ __forceinline__ v8f wmma_bf16(v16bf a, v16bf b, v8f c) {
  return __builtin_amdgcn_wmma_f32_16x16x32_bf16(false, a, false, b,
                                                 (short)0, c, false, false);
}

// ---------------------------------------------------------------- converts
__global__ void mha_cvt_kernel(const float* __restrict__ src,
                               bf16_t* __restrict__ dst, int n) {
  int i = blockIdx.x * blockDim.x + threadIdx.x;
  if (i < n) dst[i] = f32_to_bf16(src[i]);
}

// Wq[h][d][e] (fp32) -> WqT[h][e][d] (bf16)
__global__ void mha_cvtT_kernel(const float* __restrict__ src,
                                bf16_t* __restrict__ dst) {
  int i = blockIdx.x * blockDim.x + threadIdx.x;   // over H*D*HD exactly
  int h   = i / (D_ * HD_);
  int rem = i - h * (D_ * HD_);
  int d   = rem / HD_;
  int e   = rem - d * HD_;
  dst[((size_t)h * HD_ + e) * D_ + d] = f32_to_bf16(src[i]);
}

// ------------------------------------------------------------- QKV GEMMs
// Register-blocked 32x64 per wave (2 M-tiles x 4 N-tiles = full HD width).
// z=0: Q[b][h][s][e]  z=1: K[b][h][s][e]  z=2: Vt[b][h][e][s]
__global__ __launch_bounds__(32) void mha_qkv_kernel(
    const bf16_t* __restrict__ xb,  const bf16_t* __restrict__ WqT,
    const bf16_t* __restrict__ WkT, const bf16_t* __restrict__ WvT,
    bf16_t* __restrict__ Q, bf16_t* __restrict__ Kc, bf16_t* __restrict__ Vt) {
  const int lane = threadIdx.x;
  const int row0 = blockIdx.x << 5;           // 32 rows per wave
  const int bh   = blockIdx.y;
  const int b    = bh / H_, h = bh - b * H_;
  const int z    = blockIdx.z;
  const bf16_t* A = xb + (size_t)b * S_ * D_;
  const bf16_t* W = (z == 0 ? WqT : (z == 1 ? WkT : WvT)) + (size_t)h * HD_ * D_;

  v8f c[2][4] = {};
  for (int k = 0; k < D_; k += 32) {
    const v16bf a0 = load_A(A, D_, row0,      k);
    const v16bf a1 = load_A(A, D_, row0 + 16, k);
#pragma unroll
    for (int nt = 0; nt < 4; ++nt) {
      const v16bf bm = load_B(W, D_, nt << 4, k);
      c[0][nt] = wmma_bf16(a0, bm, c[0][nt]);
      c[1][nt] = wmma_bf16(a1, bm, c[1][nt]);
    }
  }
  const int rbase = (lane & 16) ? 8 : 0;
  const int colh  = lane & 15;
  const size_t base = (size_t)bh * S_ * HD_;
  if (z == 2) {                       // transposed store: Vt[e][s]
    bf16_t* dst = Vt + base;
#pragma unroll
    for (int mt = 0; mt < 2; ++mt)
#pragma unroll
      for (int nt = 0; nt < 4; ++nt)
#pragma unroll
        for (int r = 0; r < 8; ++r)
          dst[(size_t)((nt << 4) + colh) * S_ + row0 + (mt << 4) + rbase + r] =
              f32_to_bf16(c[mt][nt][r]);
  } else {
    bf16_t* dst = (z == 0 ? Q : Kc) + base;
#pragma unroll
    for (int mt = 0; mt < 2; ++mt)
#pragma unroll
      for (int nt = 0; nt < 4; ++nt)
#pragma unroll
        for (int r = 0; r < 8; ++r)
          dst[(size_t)(row0 + (mt << 4) + rbase + r) * HD_ + (nt << 4) + colh] =
              f32_to_bf16(c[mt][nt][r]);
  }
}

// ---------------------------------------------------- flash attention core
__global__ __launch_bounds__(128) void mha_flash_kernel(
    const bf16_t* __restrict__ Q, const bf16_t* __restrict__ Kc,
    const bf16_t* __restrict__ Vt, bf16_t* __restrict__ O) {
  __shared__ __align__(16) bf16_t Pl[4][16][32];   // P staging per wave
  const int lane = threadIdx.x & 31;
  const int w    = threadIdx.x >> 5;
  const int b    = blockIdx.z, h = blockIdx.y;
  const int row0 = blockIdx.x * 64 + w * 16;
  const size_t bh = (size_t)b * H_ + h;
  const bf16_t* Qb = Q  + bh * S_ * HD_;
  const bf16_t* Kb = Kc + bh * S_ * HD_;
  const bf16_t* Vb = Vt + bh * HD_ * S_;

  const v16bf aq0 = load_A(Qb, HD_, row0, 0);
  const v16bf aq1 = load_A(Qb, HD_, row0, 32);

  v8f o0 = {}, o1 = {}, o2 = {}, o3 = {};
  float m_s[8], l_s[8];
#pragma unroll
  for (int r = 0; r < 8; ++r) { m_s[r] = -__builtin_inff(); l_s[r] = 0.f; }

  const int rbase = (lane & 16) ? 8 : 0;
  const int colh  = lane & 15;
  const int nblk  = (row0 + 16 + 31) >> 5;          // causal: cols <= row0+15

  for (int jb = 0; jb < nblk; ++jb) {
    const int t = jb << 5;
    if (jb + 1 < nblk) {                            // global_prefetch next K
      __builtin_prefetch(Kb + (size_t)(t + 32) * HD_, 0, 1);
      __builtin_prefetch(Vb + (size_t)colh * S_ + t + 32, 0, 1);
    }
    // ---- scores S = Q K^T (two 16-col tiles, K-dim 64) ----
    v8f s0 = {}, s1 = {};
    s0 = wmma_bf16(aq0, load_B(Kb, HD_, t,      0), s0);
    s0 = wmma_bf16(aq1, load_B(Kb, HD_, t,     32), s0);
    s1 = wmma_bf16(aq0, load_B(Kb, HD_, t + 16, 0), s1);
    s1 = wmma_bf16(aq1, load_B(Kb, HD_, t + 16,32), s1);

    // ---- causal mask + online softmax (per-row butterflies in half-wave) --
#pragma unroll
    for (int r = 0; r < 8; ++r) {
      const int grow = row0 + rbase + r;
      float v0 = s0[r] * SCALE_;
      float v1 = s1[r] * SCALE_;
      if (t + colh      > grow) v0 = -1e30f;
      if (t + 16 + colh > grow) v1 = -1e30f;
      float mx = fmaxf(v0, v1);
      mx = fmaxf(mx, __shfl_xor(mx, 1, 32));
      mx = fmaxf(mx, __shfl_xor(mx, 2, 32));
      mx = fmaxf(mx, __shfl_xor(mx, 4, 32));
      mx = fmaxf(mx, __shfl_xor(mx, 8, 32));
      const float mnew  = fmaxf(m_s[r], mx);
      const float alpha = __expf(m_s[r] - mnew);
      const float p0 = __expf(v0 - mnew);
      const float p1 = __expf(v1 - mnew);
      float rs = p0 + p1;
      rs += __shfl_xor(rs, 1, 32);
      rs += __shfl_xor(rs, 2, 32);
      rs += __shfl_xor(rs, 4, 32);
      rs += __shfl_xor(rs, 8, 32);
      l_s[r] = l_s[r] * alpha + rs;
      m_s[r] = mnew;
      o0[r] *= alpha; o1[r] *= alpha; o2[r] *= alpha; o3[r] *= alpha;
      Pl[w][rbase + r][colh]      = f32_to_bf16(p0);
      Pl[w][rbase + r][colh + 16] = f32_to_bf16(p1);
    }
    asm volatile("s_wait_dscnt 0" ::: "memory");    // LDS RAW across lanes

    // ---- O += P V  (P from LDS as A-frag; V^T is the BT form) ----
    const v16bf pf = load_A(&Pl[w][0][0], 32, 0, 0);
    o0 = wmma_bf16(pf, load_B(Vb, S_,  0, t), o0);
    o1 = wmma_bf16(pf, load_B(Vb, S_, 16, t), o1);
    o2 = wmma_bf16(pf, load_B(Vb, S_, 32, t), o2);
    o3 = wmma_bf16(pf, load_B(Vb, S_, 48, t), o3);
  }

  // ---- normalize + store concat-head layout O[b][s][h*HD+e] ----
  bf16_t* Ob = O + ((size_t)b * S_ + row0) * D_ + h * HD_;
#pragma unroll
  for (int r = 0; r < 8; ++r) {
    const float inv = 1.0f / l_s[r];
    bf16_t* dst = Ob + (size_t)(rbase + r) * D_;
    dst[ 0 + colh] = f32_to_bf16(o0[r] * inv);
    dst[16 + colh] = f32_to_bf16(o1[r] * inv);
    dst[32 + colh] = f32_to_bf16(o2[r] * inv);
    dst[48 + colh] = f32_to_bf16(o3[r] * inv);
  }
}

// ------------------------------------------------- output projection + bias
// Register-blocked 32x64 per wave (2 M-tiles x 4 N-tiles).
__global__ __launch_bounds__(32) void mha_proj_kernel(
    const bf16_t* __restrict__ O, const bf16_t* __restrict__ Wpb,
    const float* __restrict__ bp, float* __restrict__ out) {
  const int lane = threadIdx.x;
  const int n0   = blockIdx.x << 6;   // 64 cols
  const int row0 = blockIdx.y << 5;   // 32 rows
  v8f c[2][4] = {};
  for (int k = 0; k < D_; k += 32) {
    const v16bf a0 = load_A(O, D_, row0,      k);
    const v16bf a1 = load_A(O, D_, row0 + 16, k);
#pragma unroll
    for (int nt = 0; nt < 4; ++nt) {
      const v16bf bm = load_B(Wpb, D_, n0 + (nt << 4), k);  // BT = Wp rows
      c[0][nt] = wmma_bf16(a0, bm, c[0][nt]);
      c[1][nt] = wmma_bf16(a1, bm, c[1][nt]);
    }
  }
  const int rbase = (lane & 16) ? 8 : 0;
  const int colh  = lane & 15;
#pragma unroll
  for (int nt = 0; nt < 4; ++nt) {
    const float bias = bp[n0 + (nt << 4) + colh];
#pragma unroll
    for (int mt = 0; mt < 2; ++mt)
#pragma unroll
      for (int r = 0; r < 8; ++r)
        out[(size_t)(row0 + (mt << 4) + rbase + r) * D_ + n0 + (nt << 4) + colh] =
            c[mt][nt][r] + bias;
  }
}

extern "C" void kernel_launch(void* const* d_in, const int* in_sizes, int n_in,
                              void* d_out, int out_size, void* d_ws, size_t ws_size,
                              hipStream_t stream) {
  (void)in_sizes; (void)n_in; (void)out_size; (void)ws_size;
  const float* x  = (const float*)d_in[0];
  const float* Wq = (const float*)d_in[1];
  const float* Wk = (const float*)d_in[2];
  const float* Wv = (const float*)d_in[3];
  const float* Wp = (const float*)d_in[4];
  const float* bp = (const float*)d_in[5];
  float* out = (float*)d_out;

  bf16_t* wsb = (bf16_t*)d_ws;
  size_t off = 0;
  bf16_t* xb  = wsb + off; off += (size_t)B_ * S_ * D_;
  bf16_t* WqT = wsb + off; off += (size_t)H_ * HD_ * D_;
  bf16_t* WkT = wsb + off; off += (size_t)H_ * HD_ * D_;
  bf16_t* WvT = wsb + off; off += (size_t)H_ * HD_ * D_;
  bf16_t* Wpb = wsb + off; off += (size_t)D_ * D_;
  bf16_t* Qw  = wsb + off; off += (size_t)B_ * H_ * S_ * HD_;
  bf16_t* Kw  = wsb + off; off += (size_t)B_ * H_ * S_ * HD_;
  bf16_t* Vtw = wsb + off; off += (size_t)B_ * H_ * HD_ * S_;
  bf16_t* Ow  = wsb + off; off += (size_t)B_ * S_ * D_;

  mha_cvt_kernel <<<(B_ * S_ * D_) / 256, 256, 0, stream>>>(x, xb, B_ * S_ * D_);
  mha_cvtT_kernel<<<(H_ * D_ * HD_) / 256, 256, 0, stream>>>(Wq, WqT);
  mha_cvtT_kernel<<<(H_ * D_ * HD_) / 256, 256, 0, stream>>>(Wk, WkT);
  mha_cvtT_kernel<<<(H_ * D_ * HD_) / 256, 256, 0, stream>>>(Wv, WvT);
  mha_cvt_kernel <<<(D_ * D_) / 256, 256, 0, stream>>>(Wp, Wpb, D_ * D_);

  mha_qkv_kernel <<<dim3(S_ / 32, B_ * H_, 3), 32, 0, stream>>>(
      xb, WqT, WkT, WvT, Qw, Kw, Vtw);
  mha_flash_kernel<<<dim3(S_ / 64, H_, B_), 128, 0, stream>>>(Qw, Kw, Vtw, Ow);
  mha_proj_kernel <<<dim3(D_ / 64, (B_ * S_) / 32), 32, 0, stream>>>(Ow, Wpb, bp, out);
}
